// MultiHeadAttention_24970939858953
// MI455X (gfx1250) — matrix operand order repaired and checked by reading, not compiled, not run on hardware
//
#include <hip/hip_runtime.h>

// ---------------- problem constants ----------------
#define BB 2
#define SS 2048
#define DD 768
#define HH 12
#define DK 64
// ---------------------------------------------------

typedef __attribute__((ext_vector_type(16))) __bf16        v16bf;
typedef __attribute__((ext_vector_type(4)))  __bf16        bf16x4;
typedef __attribute__((ext_vector_type(8)))  float         v8f;
typedef __attribute__((ext_vector_type(4)))  unsigned int  u32x4;
typedef __attribute__((ext_vector_type(2)))  unsigned int  u32x2;

union ABFrag { u32x4 u4[2]; v16bf v; };

__device__ __forceinline__ unsigned short f2bf(float f) {
  return __builtin_bit_cast(unsigned short, static_cast<__bf16>(f));
}

// CDNA5 async memory->LDS copy (ASYNCcnt); LDS addr = low 32 bits of generic ptr
__device__ __forceinline__ void async_load_b128(const void* gptr, void* lptr) {
  unsigned lds = (unsigned)(unsigned long long)lptr;
  asm volatile("global_load_async_to_lds_b128 %0, %1, off"
               :: "v"(lds), "v"(gptr) : "memory");
}
__device__ __forceinline__ void wait_async0() {
  asm volatile("s_wait_asynccnt 0" ::: "memory");
}

// ---------------- f32 -> bf16 convert (vectorized, native cvt) ----------------
__global__ __launch_bounds__(256) void cvt_f32_bf16(const float* __restrict__ s,
                                                    unsigned short* __restrict__ d,
                                                    int n4) {
  int i = blockIdx.x * 256 + threadIdx.x;
  if (i >= n4) return;
  float4 f = ((const float4*)s)[i];
  bf16x4 o = { (__bf16)f.x, (__bf16)f.y, (__bf16)f.z, (__bf16)f.w };
  ((bf16x4*)d)[i] = o;
}

// ---------------- mask (B,S,S) int -> packed bits ----------------
// 32MB of int mask becomes 1MB of bits (L2-resident for the attention loop).
__global__ __launch_bounds__(256) void mask_pack(const int* __restrict__ m,
                                                 unsigned int* __restrict__ mb,
                                                 int n) {
  int i = blockIdx.x * 256 + threadIdx.x;     // one int per thread, coalesced
  if (i >= n) return;
  int lane = threadIdx.x & 31;
  unsigned int bits = (m[i] != 0) ? (1u << lane) : 0u;
#pragma unroll
  for (int off = 1; off < 32; off <<= 1) bits |= __shfl_xor(bits, off, 32);
  if (lane == 0) mb[i >> 5] = bits;
}

// ---------------- WMMA GEMM: C = A(MxK) @ W(NxK)^T + bias ----------------
// MODE 0: bf16 head-split (B,H,S,DK) out. MODE 1: f32 (B,S,D) out.
// Tile 128x128x32; 8 waves as 4(M) x 2(N); wave tile 32x64.
// Double-buffered LDS; next slab's async copy is in flight under current WMMAs.
template<int MODE>
__global__ __launch_bounds__(256) void gemm_bf16(const unsigned short* __restrict__ A,
                                                 const unsigned short* __restrict__ W,
                                                 const float* __restrict__ bias,
                                                 void* __restrict__ Cout,
                                                 int K) {
  __shared__ alignas(16) unsigned short lA[2][128 * 32];
  __shared__ alignas(16) unsigned short lB[2][128 * 32];

  const int t   = threadIdx.x;
  const int wid = t >> 5, lane = t & 31;
  const int hv  = lane >> 4, r = lane & 15;
  const int waveM = wid >> 1, waveN = wid & 1;
  const int n0 = blockIdx.x * 128;
  const int m0 = blockIdx.y * 128;

  auto stage = [&](int kt, int pb) {
#pragma unroll
    for (int i = 0; i < 2; ++i) {
      int s   = t * 2 + i;
      int row = s >> 2, c4 = s & 3;
      async_load_b128(&A[(size_t)(m0 + row) * K + kt + c4 * 8],
                      &lA[pb][row * 32 + c4 * 8]);
      async_load_b128(&W[(size_t)(n0 + row) * K + kt + c4 * 8],
                      &lB[pb][row * 32 + c4 * 8]);
    }
  };

  v8f acc[2][4];
#pragma unroll
  for (int mt = 0; mt < 2; ++mt)
#pragma unroll
    for (int nt = 0; nt < 4; ++nt)
#pragma unroll
      for (int e = 0; e < 8; ++e) acc[mt][nt][e] = 0.f;

  stage(0, 0);
  const int nk = K / 32;
  for (int ki = 0; ki < nk; ++ki) {
    wait_async0();            // tile ki resident in LDS
    __syncthreads();          // ...in all waves; prior reads of other buffer done
    const int cur = ki & 1;
    if (ki + 1 < nk) stage((ki + 1) * 32, cur ^ 1);   // overlap with compute

    ABFrag a[2], b[4];
#pragma unroll
    for (int mt = 0; mt < 2; ++mt) {              // A frag: k = {8h..}, {16+8h..}
      int row = waveM * 32 + mt * 16 + r;
      a[mt].u4[0] = *(const u32x4*)&lA[cur][row * 32 + hv * 8];
      a[mt].u4[1] = *(const u32x4*)&lA[cur][row * 32 + 16 + hv * 8];
    }
#pragma unroll
    for (int nt = 0; nt < 4; ++nt) {              // B frag: k = {8h..}, {16+8h..}
      int col = waveN * 64 + nt * 16 + r;
      b[nt].u4[0] = *(const u32x4*)&lB[cur][col * 32 + hv * 8];
      b[nt].u4[1] = *(const u32x4*)&lB[cur][col * 32 + 16 + hv * 8];
    }
#pragma unroll
    for (int mt = 0; mt < 2; ++mt)
#pragma unroll
      for (int nt = 0; nt < 4; ++nt)
        acc[mt][nt] = __builtin_amdgcn_wmma_f32_16x16x32_bf16(
            false, a[mt].v, false, b[nt].v, (short)0, acc[mt][nt], false, false);
  }

#pragma unroll
  for (int mt = 0; mt < 2; ++mt)
#pragma unroll
    for (int nt = 0; nt < 4; ++nt)
#pragma unroll
      for (int e = 0; e < 8; ++e) {
        int gm = m0 + waveM * 32 + mt * 16 + e + 8 * hv;
        int gn = n0 + waveN * 64 + nt * 16 + r;
        float val = acc[mt][nt][e] + bias[gn];
        if (MODE == 0) {
          int bb = gm >> 11, ss = gm & (SS - 1);
          int hh = gn >> 6, dk = gn & (DK - 1);
          ((unsigned short*)Cout)[(((size_t)bb * HH + hh) * SS + ss) * DK + dk] = f2bf(val);
        } else {
          ((float*)Cout)[(size_t)gm * DD + gn] = val;
        }
      }
}

// ---------------- flash attention ----------------
// 8 waves x 16 q-rows per block; 64-key tiles; online softmax; WMMA bf16 matmuls.
// Double-buffered K/V staging pipelined under the softmax + WMMA work.
__global__ __launch_bounds__(256) void flash_attn(const unsigned short* __restrict__ Qh,
                                                  const unsigned short* __restrict__ Kh,
                                                  const unsigned short* __restrict__ Vh,
                                                  const unsigned int* __restrict__ mb,
                                                  unsigned short* __restrict__ Ctx) {
  __shared__ alignas(16) unsigned short lK[2][64 * 64];    // [key][dk]
  __shared__ alignas(16) unsigned short lVt[2][64 * 64];   // [dk][key]
  __shared__ alignas(16) unsigned short lP[8 * 16 * 64];   // per-wave P tiles

  const int t   = threadIdx.x;
  const int wid = t >> 5, lane = t & 31;
  const int hv  = lane >> 4, r = lane & 15;
  const int bh  = blockIdx.y;
  const int b   = bh / HH, h = bh % HH;
  const size_t headOff = (size_t)bh * SS * DK;
  const unsigned short* Qp = Qh + headOff;
  const unsigned short* Kp = Kh + headOff;
  const unsigned short* Vp = Vh + headOff;
  const unsigned int* mbq = mb + (size_t)b * SS * (SS / 32);
  const int m0 = blockIdx.x * 128 + wid * 16;

  auto stageKV = [&](int kt0, int pb) {
#pragma unroll
    for (int i = 0; i < 2; ++i) {
      int s = t + i * 256;
      int row = s >> 3, c = s & 7;
      async_load_b128(&Kp[(size_t)(kt0 + row) * DK + c * 8],
                      &lK[pb][row * 64 + c * 8]);
      u32x4 dv = *(const u32x4*)&Vp[(size_t)(kt0 + row) * DK + c * 8];
#pragma unroll
      for (int j = 0; j < 8; ++j) {
        unsigned int w = dv[j >> 1];
        lVt[pb][(c * 8 + j) * 64 + row] =
            (unsigned short)((j & 1) ? (w >> 16) : (w & 0xffffu));
      }
    }
  };

  // Q fragments stay in registers for the whole kernel
  ABFrag qa[2];
#pragma unroll
  for (int ks = 0; ks < 2; ++ks) {
    const unsigned short* qrow = Qp + (size_t)(m0 + r) * DK + ks * 32;
    qa[ks].u4[0] = *(const u32x4*)&qrow[hv * 8];
    qa[ks].u4[1] = *(const u32x4*)&qrow[16 + hv * 8];
  }

  v8f oacc[4];
  float mx[8], sm[8];
#pragma unroll
  for (int dt = 0; dt < 4; ++dt)
#pragma unroll
    for (int e = 0; e < 8; ++e) oacc[dt][e] = 0.f;
#pragma unroll
  for (int e = 0; e < 8; ++e) { mx[e] = -1e30f; sm[e] = 0.f; }

  unsigned short* lPw = lP + wid * (16 * 64);

  stageKV(0, 0);
  for (int it = 0; it < SS / 64; ++it) {
    const int kt0 = it * 64;
    wait_async0();            // K tile(it) resident
    __syncthreads();          // V(it) ds_stores visible; prior buffer reads done
    const int cur = it & 1;
    if (kt0 + 64 < SS) stageKV(kt0 + 64, cur ^ 1);    // overlap with compute

    // S = Q @ K^T
    v8f sc[4];
#pragma unroll
    for (int nt = 0; nt < 4; ++nt)
#pragma unroll
      for (int e = 0; e < 8; ++e) sc[nt][e] = 0.f;
#pragma unroll
    for (int ks = 0; ks < 2; ++ks)
#pragma unroll
      for (int nt = 0; nt < 4; ++nt) {
        ABFrag bk;
        const unsigned short* kb = &lK[cur][(nt * 16 + r) * 64 + ks * 32 + hv * 8];
        bk.u4[0] = *(const u32x4*)&kb[0];
        bk.u4[1] = *(const u32x4*)&kb[16];
        sc[nt] = __builtin_amdgcn_wmma_f32_16x16x32_bf16(
            false, qa[ks].v, false, bk.v, (short)0, sc[nt], false, false);
      }

    // scale + packed-bit mask + tile row-max (8B of mask per row, L2-resident)
    float tmax[8];
#pragma unroll
    for (int e = 0; e < 8; ++e) {
      int gq = m0 + 8 * hv + e;
      u32x2 w2 = *(const u32x2*)&mbq[(size_t)gq * (SS / 32) + (kt0 >> 5)];
      float tm = -1e30f;
#pragma unroll
      for (int nt = 0; nt < 4; ++nt) {
        unsigned int wv = (nt & 2) ? w2.y : w2.x;
        float sv = sc[nt][e] * 0.125f;                 // 1/sqrt(64)
        if (((wv >> ((nt & 1) * 16 + r)) & 1u) == 0u) sv = -1e9f;
        sc[nt][e] = sv;
        tm = fmaxf(tm, sv);
      }
      tmax[e] = tm;
    }
#pragma unroll
    for (int e = 0; e < 8; ++e)
#pragma unroll
      for (int off = 8; off >= 1; off >>= 1)
        tmax[e] = fmaxf(tmax[e], __shfl_xor(tmax[e], off, 32));

    // online softmax update
    float alpha[8], tsum[8];
#pragma unroll
    for (int e = 0; e < 8; ++e) {
      float mnew = fmaxf(mx[e], tmax[e]);
      alpha[e] = __builtin_amdgcn_exp2f((mx[e] - mnew) * 1.44269504f);
      mx[e] = mnew;
      tsum[e] = 0.f;
    }
#pragma unroll
    for (int nt = 0; nt < 4; ++nt)
#pragma unroll
      for (int e = 0; e < 8; ++e) {
        float p = __builtin_amdgcn_exp2f((sc[nt][e] - mx[e]) * 1.44269504f);
        tsum[e] += p;
        lPw[(e + 8 * hv) * 64 + nt * 16 + r] = f2bf(p);
      }
#pragma unroll
    for (int e = 0; e < 8; ++e) {
#pragma unroll
      for (int off = 8; off >= 1; off >>= 1)
        tsum[e] += __shfl_xor(tsum[e], off, 32);
      sm[e] = sm[e] * alpha[e] + tsum[e];
    }
#pragma unroll
    for (int dt = 0; dt < 4; ++dt)
#pragma unroll
      for (int e = 0; e < 8; ++e) oacc[dt][e] *= alpha[e];

    __syncthreads();          // P tiles visible across lanes

    // O += P @ V
#pragma unroll
    for (int ks = 0; ks < 2; ++ks) {
      ABFrag pa;
      const unsigned short* pr = lPw + (size_t)r * 64 + ks * 32;
      pa.u4[0] = *(const u32x4*)&pr[hv * 8];
      pa.u4[1] = *(const u32x4*)&pr[16 + hv * 8];
#pragma unroll
      for (int dt = 0; dt < 4; ++dt) {
        ABFrag vb;
        const unsigned short* vrow = &lVt[cur][(dt * 16 + r) * 64 + ks * 32 + hv * 8];
        vb.u4[0] = *(const u32x4*)&vrow[0];
        vb.u4[1] = *(const u32x4*)&vrow[16];
        oacc[dt] = __builtin_amdgcn_wmma_f32_16x16x32_bf16(
            false, pa.v, false, vb.v, (short)0, oacc[dt], false, false);
      }
    }
  }

  // normalize and write context (B,S,D) bf16
#pragma unroll
  for (int e = 0; e < 8; ++e)
    sm[e] = (sm[e] > 0.f) ? __builtin_amdgcn_rcpf(sm[e]) : 0.f;
#pragma unroll
  for (int dt = 0; dt < 4; ++dt)
#pragma unroll
    for (int e = 0; e < 8; ++e) {
      int srow = m0 + e + 8 * hv;
      int dk = dt * 16 + r;
      Ctx[((size_t)b * SS + srow) * DD + h * DK + dk] = f2bf(oacc[dt][e] * sm[e]);
    }
}

// ---------------- host orchestration ----------------
extern "C" void kernel_launch(void* const* d_in, const int* in_sizes, int n_in,
                              void* d_out, int out_size, void* d_ws, size_t ws_size,
                              hipStream_t stream) {
  const float* q   = (const float*)d_in[0];
  const float* k   = (const float*)d_in[1];
  const float* v   = (const float*)d_in[2];
  const int*  mask = (const int*)d_in[3];
  const float* Wq  = (const float*)d_in[4];
  const float* bq  = (const float*)d_in[5];
  const float* Wk  = (const float*)d_in[6];
  const float* bk  = (const float*)d_in[7];
  const float* Wv  = (const float*)d_in[8];
  const float* bv  = (const float*)d_in[9];
  const float* Wo  = (const float*)d_in[10];
  const float* bo  = (const float*)d_in[11];

  const size_t nEmb = (size_t)BB * SS * DD;   // 3,145,728
  const size_t nW   = (size_t)DD * DD;        //   589,824
  unsigned short* ws    = (unsigned short*)d_ws;
  unsigned short* q_bf  = ws;
  unsigned short* k_bf  = q_bf + nEmb;
  unsigned short* v_bf  = k_bf + nEmb;
  unsigned short* Wq_bf = v_bf + nEmb;
  unsigned short* Wk_bf = Wq_bf + nW;
  unsigned short* Wv_bf = Wk_bf + nW;
  unsigned short* Wo_bf = Wv_bf + nW;
  unsigned short* Qh    = Wo_bf + nW;         // (B,H,S,DK) bf16
  unsigned short* Kh    = Qh + nEmb;
  unsigned short* Vh    = Kh + nEmb;
  unsigned short* Ctx   = Vh + nEmb;          // (B,S,D) bf16
  unsigned int*  mbits  = (unsigned int*)(Ctx + nEmb);  // B*S*S/32 words (1MB)

  // f32 -> bf16
  cvt_f32_bf16<<<(int)(nEmb / 1024), 256, 0, stream>>>(q, q_bf, (int)(nEmb / 4));
  cvt_f32_bf16<<<(int)(nEmb / 1024), 256, 0, stream>>>(k, k_bf, (int)(nEmb / 4));
  cvt_f32_bf16<<<(int)(nEmb / 1024), 256, 0, stream>>>(v, v_bf, (int)(nEmb / 4));
  cvt_f32_bf16<<<(int)(nW / 1024), 256, 0, stream>>>(Wq, Wq_bf, (int)(nW / 4));
  cvt_f32_bf16<<<(int)(nW / 1024), 256, 0, stream>>>(Wk, Wk_bf, (int)(nW / 4));
  cvt_f32_bf16<<<(int)(nW / 1024), 256, 0, stream>>>(Wv, Wv_bf, (int)(nW / 4));
  cvt_f32_bf16<<<(int)(nW / 1024), 256, 0, stream>>>(Wo, Wo_bf, (int)(nW / 4));

  // pack 32MB int mask into 1MB of bits (single streaming pass)
  const int nMask = BB * SS * SS;             // 8,388,608
  mask_pack<<<nMask / 256, 256, 0, stream>>>(mask, mbits, nMask);

  // projections
  dim3 gg(DD / 128, (BB * SS) / 128);         // (6, 32)
  gemm_bf16<0><<<gg, 256, 0, stream>>>(q_bf, Wq_bf, bq, Qh, DD);
  gemm_bf16<0><<<gg, 256, 0, stream>>>(k_bf, Wk_bf, bk, Kh, DD);
  gemm_bf16<0><<<gg, 256, 0, stream>>>(v_bf, Wv_bf, bv, Vh, DD);

  // attention
  flash_attn<<<dim3(SS / 128, BB * HH), 256, 0, stream>>>(Qh, Kh, Vh, mbits, Ctx);

  // output projection -> f32
  gemm_bf16<1><<<gg, 256, 0, stream>>>(Ctx, Wo_bf, bo, d_out, DD);
}
